// HopfieldAttentionBackbone_70068096467991
// MI455X (gfx1250) — matrix operand order repaired and checked by reading, not compiled, not run on hardware
//
#include <hip/hip_runtime.h>
#include <hip/hip_bf16.h>

// ---------------------------------------------------------------------------
// Hopfield attention backbone for MI455X (gfx1250).
// All GEMMs: bf16 WMMA (v_wmma_f32_16x16x32_bf16), f32 accumulate.
// LDS staging via GLOBAL_LOAD_ASYNC_TO_LDS_B128 (ASYNCcnt), double-buffered,
// one barrier per K-step. Block tile 256x128, wave tile 64x64.
// ---------------------------------------------------------------------------

typedef __bf16 v16bf __attribute__((ext_vector_type(16)));
typedef float  v8f   __attribute__((ext_vector_type(8)));

union FragU {
    uint4          u[2];
    v16bf          v;
    unsigned short s[16];
};

__device__ __forceinline__ unsigned short f2bf(float f) {
    union { float f; unsigned u; } v; v.f = f;
    unsigned u = v.u;
    unsigned r = (u + 0x7FFFu + ((u >> 16) & 1u)) >> 16;   // RNE
    return (unsigned short)r;
}

#if __has_builtin(__builtin_amdgcn_global_load_async_to_lds_b128)
#define ASYNC_LDS 1
typedef int v4i __attribute__((vector_size(16)));
typedef __attribute__((address_space(1))) v4i as1_v4i;
typedef __attribute__((address_space(3))) v4i as3_v4i;
#endif

__device__ __forceinline__ void async_ld16(const unsigned short* g, unsigned short* l) {
#if defined(ASYNC_LDS)
    __builtin_amdgcn_global_load_async_to_lds_b128(
        (as1_v4i*)(void*)g, (as3_v4i*)(void*)l, 0, 0);
#else
    *reinterpret_cast<uint4*>(l) = *reinterpret_cast<const uint4*>(g);
#endif
}

__device__ __forceinline__ void wait_async0() {
#if defined(ASYNC_LDS)
#if __has_builtin(__builtin_amdgcn_s_wait_asynccnt)
    __builtin_amdgcn_s_wait_asynccnt(0);
#else
    asm volatile("s_wait_asynccnt 0x0" ::: "memory");
#endif
#endif
}

// ---------------------------------------------------------------------------
// Generic NT GEMM:  C[m,n] = alpha * sum_k A[m,k] * B[n,k] + bias[n]
//   A  : bf16 bits, row-major [M,K], lda = K
//   B  : bf16 (BF32=false, async-staged) or f32 (BF32=true, reg-staged+cvt)
//   Cf : optional f32 out [M,N]; Cb : optional bf16 out [M,N]
//   Batched via gridDim.z with element strides sA, sB, sC.
// Block: 256 threads (8 waves, wave32). Tile 256x128, BK=32, double-buffered
// LDS (pad stride 40 halves -> conflict-free ds_load_b128 fragments).
// Waves 4(M) x 2(N); wave tile 64x64 = 4x4 WMMA tiles (16 WMMA / K-step).
// M % 256 == 0, K % 32 == 0 required.
// ---------------------------------------------------------------------------
template <bool BF32>
__global__ __launch_bounds__(256) void gemm_wmma_nt(
    const unsigned short* __restrict__ A,
    const void* __restrict__ Bp,
    const float* __restrict__ bias,
    const float* __restrict__ alpha_ptr, float alpha_const,
    float* __restrict__ Cf,
    unsigned short* __restrict__ Cb,
    int M, int N, int K,
    long long sA, long long sB, long long sC)
{
    __shared__ __align__(16) unsigned short As[2][256 * 40];   // 2 x 20 KB
    __shared__ __align__(16) unsigned short Bs[2][128 * 40];   // 2 x 10 KB

    const int tid   = threadIdx.x;
    const int lane  = tid & 31;
    const int wid   = tid >> 5;
    const int waveM = wid >> 1;        // 0..3
    const int waveN = wid & 1;         // 0..1
    const int half  = lane >> 4;       // 0/1
    const int l15   = lane & 15;

    const int bM = blockIdx.y * 256;
    const int bN = blockIdx.x * 128;

    A += (long long)blockIdx.z * sA;
    const unsigned short* Bb = BF32 ? nullptr : (const unsigned short*)Bp + (long long)blockIdx.z * sB;
    const float*          Bf = BF32 ? (const float*)Bp + (long long)blockIdx.z * sB : nullptr;

    // staging: A tile 256x32 -> 1024 x 16B chunks (4/thread),
    //          B tile 128x32 ->  512 x 16B chunks (2/thread)
    const int row0 = tid >> 2;             // 0..63
    const int cc   = (tid & 3) * 8;        // 0,8,16,24
    size_t gA[4]; int lA[4];
#pragma unroll
    for (int j = 0; j < 4; ++j) {
        gA[j] = (size_t)(bM + row0 + 64 * j) * K + cc;
        lA[j] = (row0 + 64 * j) * 40 + cc;
    }
    size_t gB[2];
#pragma unroll
    for (int j = 0; j < 2; ++j) {
        int n = bN + row0 + 64 * j;
        if (n >= N) n = N - 1;             // clamp (stores guarded later)
        gB[j] = (size_t)n * K + cc;
    }

    v8f acc[4][4];
#pragma unroll
    for (int i = 0; i < 4; ++i)
#pragma unroll
        for (int j = 0; j < 4; ++j)
#pragma unroll
            for (int r = 0; r < 8; ++r) acc[i][j][r] = 0.0f;

    // fragment LDS element offsets
    const int afo0 = (waveM * 64 + l15) * 40 + half * 8;    // + tm*16*40
    const int bfo0 = (waveN * 64 + l15) * 40 + half * 8;    // + tn*16*40

    uint4  ra[4];              // BF32 path register staging
    float4 rb[4];

    auto stage_async = [&](int buf, int k0) {
#pragma unroll
        for (int j = 0; j < 4; ++j) async_ld16(A + gA[j] + k0, &As[buf][lA[j]]);
#pragma unroll
        for (int j = 0; j < 2; ++j) async_ld16(Bb + gB[j] + k0, &Bs[buf][lA[j]]);
    };
    auto load_regs = [&](int k0) {
#pragma unroll
        for (int j = 0; j < 4; ++j)
            ra[j] = *reinterpret_cast<const uint4*>(A + gA[j] + k0);
#pragma unroll
        for (int j = 0; j < 2; ++j) {
            const float4* p = reinterpret_cast<const float4*>(Bf + gB[j] + k0);
            rb[2 * j]     = p[0];
            rb[2 * j + 1] = p[1];
        }
    };
    auto store_lds = [&](int buf) {
#pragma unroll
        for (int j = 0; j < 4; ++j)
            *reinterpret_cast<uint4*>(&As[buf][lA[j]]) = ra[j];
#pragma unroll
        for (int j = 0; j < 2; ++j) {
            union { unsigned short s[8]; uint4 u; } pk;
            float4 f0 = rb[2 * j], f1 = rb[2 * j + 1];
            pk.s[0] = f2bf(f0.x); pk.s[1] = f2bf(f0.y);
            pk.s[2] = f2bf(f0.z); pk.s[3] = f2bf(f0.w);
            pk.s[4] = f2bf(f1.x); pk.s[5] = f2bf(f1.y);
            pk.s[6] = f2bf(f1.z); pk.s[7] = f2bf(f1.w);
            *reinterpret_cast<uint4*>(&Bs[buf][lA[j]]) = pk.u;
        }
    };
    auto compute = [&](int buf) {
        const unsigned short* Asb = As[buf];
        const unsigned short* Bsb = Bs[buf];
        FragU bfrag[4];
#pragma unroll
        for (int tn = 0; tn < 4; ++tn) {
            const uint4* p = reinterpret_cast<const uint4*>(&Bsb[bfo0 + tn * 16 * 40]);
            bfrag[tn].u[0] = p[0];
            bfrag[tn].u[1] = p[2];
        }
#pragma unroll
        for (int tm = 0; tm < 4; ++tm) {
            FragU af;
            int base = afo0 + tm * 16 * 40;
            af.u[0] = *reinterpret_cast<const uint4*>(&Asb[base]);
            af.u[1] = *reinterpret_cast<const uint4*>(&Asb[base + 16]);
#pragma unroll
            for (int tn = 0; tn < 4; ++tn) {
                acc[tm][tn] = __builtin_amdgcn_wmma_f32_16x16x32_bf16(
                    false, af.v, false, bfrag[tn].v, (short)0, acc[tm][tn], false, false);
            }
        }
    };

    if (!BF32) {
        stage_async(0, 0);
        int buf = 0;
        for (int k0 = 0; k0 < K; k0 += 32) {
            wait_async0();
            __syncthreads();
            if (k0 + 32 < K) stage_async(buf ^ 1, k0 + 32);
            compute(buf);
            buf ^= 1;
        }
    } else {
        load_regs(0);
        int buf = 0;
        for (int k0 = 0; k0 < K; k0 += 32) {
            store_lds(buf);
            if (k0 + 32 < K) load_regs(k0 + 32);
            __syncthreads();
            compute(buf);
            buf ^= 1;
        }
    }

    // ---- epilogue: C tile layout (VGPR r: M = r + 8*half, N = l15) ----
    const float eff_alpha = alpha_const * (alpha_ptr ? *alpha_ptr : 1.0f);
    const long long cbase = (long long)blockIdx.z * sC;
#pragma unroll
    for (int tm = 0; tm < 4; ++tm) {
#pragma unroll
        for (int tn = 0; tn < 4; ++tn) {
            int col = bN + waveN * 64 + tn * 16 + l15;
            if (col >= N) continue;
            float b = bias ? bias[col] : 0.0f;
#pragma unroll
            for (int r = 0; r < 8; ++r) {
                int row = bM + waveM * 64 + tm * 16 + half * 8 + r;
                float v = eff_alpha * acc[tm][tn][r] + b;
                size_t idx = (size_t)(cbase + (long long)row * N + col);
                if (Cf) Cf[idx] = v;
                if (Cb) Cb[idx] = f2bf(v);
            }
        }
    }
}

// ---------------------------------------------------------------------------
// f32 -> bf16 elementwise (n % 4 == 0)
// ---------------------------------------------------------------------------
__global__ __launch_bounds__(256) void cvt_bf16(
    const float* __restrict__ in, unsigned short* __restrict__ out, long long n)
{
    long long i = ((long long)blockIdx.x * 256 + threadIdx.x) * 4;
    if (i >= n) return;
    float4 f = *reinterpret_cast<const float4*>(in + i);
    union { unsigned short s[4]; uint2 u; } pk;
    pk.s[0] = f2bf(f.x); pk.s[1] = f2bf(f.y);
    pk.s[2] = f2bf(f.z); pk.s[3] = f2bf(f.w);
    *reinterpret_cast<uint2*>(out + i) = pk.u;
}

// ---------------------------------------------------------------------------
// Embedding gather -> bf16 activations
// ---------------------------------------------------------------------------
__global__ __launch_bounds__(256) void embed_gather(
    const int* __restrict__ ids, const float* __restrict__ emb,
    unsigned short* __restrict__ xbf, int H)
{
    int t  = blockIdx.x;
    int id = ids[t];
    const float* src = emb + (size_t)id * H;
    unsigned short* dst = xbf + (size_t)t * H;
    for (int i = threadIdx.x; i < H; i += 256) dst[i] = f2bf(src[i]);
}

// ---------------------------------------------------------------------------
// Row softmax: scores f32 [rows, n] -> weights bf16
// ---------------------------------------------------------------------------
__global__ __launch_bounds__(256) void softmax_rows(
    const float* __restrict__ S, unsigned short* __restrict__ W, int n)
{
    __shared__ float red[256];
    const int row = blockIdx.x;
    const float* p = S + (size_t)row * n;
    unsigned short* o = W + (size_t)row * n;
    const int tid = threadIdx.x;

    float mx = -3.0e38f;
    for (int i = tid; i < n; i += 256) mx = fmaxf(mx, p[i]);
    red[tid] = mx; __syncthreads();
    for (int s = 128; s > 0; s >>= 1) {
        if (tid < s) red[tid] = fmaxf(red[tid], red[tid + s]);
        __syncthreads();
    }
    mx = red[0]; __syncthreads();

    float sum = 0.0f;
    for (int i = tid; i < n; i += 256) sum += __expf(p[i] - mx);
    red[tid] = sum; __syncthreads();
    for (int s = 128; s > 0; s >>= 1) {
        if (tid < s) red[tid] += red[tid + s];
        __syncthreads();
    }
    const float inv = 1.0f / red[0];

    for (int i = tid; i < n; i += 256) o[i] = f2bf(__expf(p[i] - mx) * inv);
}

// ---------------------------------------------------------------------------
// bf16 transpose [S,H] -> [H,S] per batch (gridDim.z)
// ---------------------------------------------------------------------------
__global__ __launch_bounds__(256) void transpose_bf16(
    const unsigned short* __restrict__ in, unsigned short* __restrict__ out,
    int S, int H)
{
    __shared__ unsigned short t[32][33];
    const size_t boff = (size_t)blockIdx.z * S * H;
    in  += boff;
    out += boff;
    int tx = threadIdx.x & 31;
    int ty = threadIdx.x >> 5;
    int h  = blockIdx.x * 32 + tx;
#pragma unroll
    for (int r = 0; r < 32; r += 8) {
        int s = blockIdx.y * 32 + ty + r;
        t[ty + r][tx] = in[(size_t)s * H + h];
    }
    __syncthreads();
    int s2 = blockIdx.y * 32 + tx;
#pragma unroll
    for (int r = 0; r < 32; r += 8) {
        int h2 = blockIdx.x * 32 + ty + r;
        out[(size_t)h2 * S + s2] = t[tx][ty + r];
    }
}

// ---------------------------------------------------------------------------
// Mean-pool over S:  pooled[b,h] = mean_s x[b,s,h]
// ---------------------------------------------------------------------------
__global__ __launch_bounds__(256) void mean_pool(
    const float* __restrict__ x, float* __restrict__ pooled, int S, int H)
{
    int t = blockIdx.x * 256 + threadIdx.x;   // t in [0, B*H)
    int b = t / H;
    int h = t % H;
    const float* p = x + (size_t)b * S * H + h;
    float sum = 0.0f;
    for (int s = 0; s < S; ++s) sum += p[(size_t)s * H];
    pooled[t] = sum / (float)S;
}

// ---------------------------------------------------------------------------
// Launch
// ---------------------------------------------------------------------------
extern "C" void kernel_launch(void* const* d_in, const int* in_sizes, int n_in,
                              void* d_out, int out_size, void* d_ws, size_t ws_size,
                              hipStream_t stream)
{
    (void)in_sizes; (void)n_in; (void)out_size;

    constexpr int Bz = 2, S = 2048, H = 1024, V = 50257, L = 2;
    constexpr int T  = Bz * S;                 // 4096 tokens

    const int*   ids  = (const int*)d_in[0];
    const float* emb  = (const float*)d_in[1];
    const float* Wq   = (const float*)d_in[2];
    const float* Wk   = (const float*)d_in[3];
    const float* Wv   = (const float*)d_in[4];
    const float* Wo   = (const float*)d_in[5];
    const float* bq   = (const float*)d_in[6];
    const float* bk   = (const float*)d_in[7];
    const float* bv   = (const float*)d_in[8];
    const float* bo   = (const float*)d_in[9];
    const float* beta = (const float*)d_in[10];
    const float* Wlm  = (const float*)d_in[11];
    const float* blm  = (const float*)d_in[12];

    float* out = (float*)d_out;

    // ---- workspace layout ----
    char* w = (char*)d_ws;
    size_t off = 0;
    auto alloc = [&](size_t bytes) -> char* {
        char* p = w + off;
        off = (off + bytes + 255) & ~(size_t)255;
        return p;
    };
    const size_t TH  = (size_t)T * H;                  // 4M elements
    const size_t HH2 = (size_t)L * H * H;              // per weight tensor (all layers)
    const size_t VH  = (size_t)V * H;

    unsigned short* xbf  = (unsigned short*)alloc(TH * 2);
    unsigned short* Qb   = (unsigned short*)alloc(TH * 2);
    unsigned short* Kb   = (unsigned short*)alloc(TH * 2);
    unsigned short* Vb   = (unsigned short*)alloc(TH * 2);
    unsigned short* Vt   = (unsigned short*)alloc(TH * 2);
    unsigned short* ctxb = (unsigned short*)alloc(TH * 2);
    float*          xf32 = (float*)alloc(TH * 4);
    float*          scf  = (float*)alloc((size_t)Bz * S * S * 4);
    unsigned short* wbf  = (unsigned short*)alloc((size_t)Bz * S * S * 2);
    unsigned short* Wqb  = (unsigned short*)alloc(HH2 * 2);
    unsigned short* Wkb  = (unsigned short*)alloc(HH2 * 2);
    unsigned short* Wvb  = (unsigned short*)alloc(HH2 * 2);
    unsigned short* Wob  = (unsigned short*)alloc(HH2 * 2);
    unsigned short* Wlmb = (unsigned short*)alloc(VH * 2);
    const bool cvt = (ws_size >= off) || (ws_size == 0);

    const long long sQ  = (long long)S * H;        // per-batch Q/K/V stride
    const long long sSc = (long long)S * S;        // per-batch score stride
    const dim3 blk(256);

    // 0) one-time (per call) weight conversion to bf16
    if (cvt) {
        dim3 gw((unsigned)((HH2 / 4 + 255) / 256));
        cvt_bf16<<<gw, blk, 0, stream>>>(Wq, Wqb, (long long)HH2);
        cvt_bf16<<<gw, blk, 0, stream>>>(Wk, Wkb, (long long)HH2);
        cvt_bf16<<<gw, blk, 0, stream>>>(Wv, Wvb, (long long)HH2);
        cvt_bf16<<<gw, blk, 0, stream>>>(Wo, Wob, (long long)HH2);
        dim3 gv((unsigned)((VH / 4 + 255) / 256));
        cvt_bf16<<<gv, blk, 0, stream>>>(Wlm, Wlmb, (long long)VH);
    }

    // 1) embedding gather -> bf16
    embed_gather<<<T, blk, 0, stream>>>(ids, emb, xbf, H);

    for (int l = 0; l < L; ++l) {
        const size_t wo  = (size_t)l * H * H;
        const size_t bo_ = (size_t)l * H;

        // 2) Q/K/V projections: [4096,1024] x [1024,1024]^T
        dim3 gp(H / 128, T / 256, 1);
        if (cvt) {
            gemm_wmma_nt<false><<<gp, blk, 0, stream>>>(
                xbf, Wqb + wo, bq + bo_, nullptr, 1.0f, nullptr, Qb, T, H, H, 0, 0, 0);
            gemm_wmma_nt<false><<<gp, blk, 0, stream>>>(
                xbf, Wkb + wo, bk + bo_, nullptr, 1.0f, nullptr, Kb, T, H, H, 0, 0, 0);
            gemm_wmma_nt<false><<<gp, blk, 0, stream>>>(
                xbf, Wvb + wo, bv + bo_, nullptr, 1.0f, nullptr, Vb, T, H, H, 0, 0, 0);
        } else {
            gemm_wmma_nt<true><<<gp, blk, 0, stream>>>(
                xbf, Wq + wo, bq + bo_, nullptr, 1.0f, nullptr, Qb, T, H, H, 0, 0, 0);
            gemm_wmma_nt<true><<<gp, blk, 0, stream>>>(
                xbf, Wk + wo, bk + bo_, nullptr, 1.0f, nullptr, Kb, T, H, H, 0, 0, 0);
            gemm_wmma_nt<true><<<gp, blk, 0, stream>>>(
                xbf, Wv + wo, bv + bo_, nullptr, 1.0f, nullptr, Vb, T, H, H, 0, 0, 0);
        }

        // 3) V^T per batch: [S,H] -> [H,S]
        dim3 gt(H / 32, S / 32, Bz);
        transpose_bf16<<<gt, blk, 0, stream>>>(Vb, Vt, S, H);

        // 4) scores = (beta/32) * Q K^T   [S,S] per batch
        dim3 gs(S / 128, S / 256, Bz);
        gemm_wmma_nt<false><<<gs, blk, 0, stream>>>(
            Qb, Kb, nullptr, beta + l, 1.0f / 32.0f, scf, nullptr,
            S, S, H, sQ, sQ, sSc);

        // 5) softmax rows -> bf16 weights
        softmax_rows<<<Bz * S, blk, 0, stream>>>(scf, wbf, S);

        // 6) ctx = weights @ V  (NT with V^T): [S,S] x [H,S]^T per batch
        dim3 gc(H / 128, S / 256, Bz);
        gemm_wmma_nt<false><<<gc, blk, 0, stream>>>(
            wbf, Vt, nullptr, nullptr, 1.0f, nullptr, ctxb,
            S, H, S, sSc, sQ, sQ);

        // 7) out projection: writes next-layer bf16 + f32 (for pooling)
        if (cvt) {
            gemm_wmma_nt<false><<<gp, blk, 0, stream>>>(
                ctxb, Wob + wo, bo + bo_, nullptr, 1.0f, xf32, xbf, T, H, H, 0, 0, 0);
        } else {
            gemm_wmma_nt<true><<<gp, blk, 0, stream>>>(
                ctxb, Wo + wo, bo + bo_, nullptr, 1.0f, xf32, xbf, T, H, H, 0, 0, 0);
        }
    }

    // 8) LM head: logits = x @ Wlm^T + blm  -> d_out
    dim3 gl((V + 127) / 128, T / 256, 1);
    if (cvt) {
        gemm_wmma_nt<false><<<gl, blk, 0, stream>>>(
            xbf, Wlmb, blm, nullptr, 1.0f, out, nullptr, T, V, H, 0, 0, 0);
    } else {
        gemm_wmma_nt<true><<<gl, blk, 0, stream>>>(
            xbf, Wlm, blm, nullptr, 1.0f, out, nullptr, T, V, H, 0, 0, 0);
    }

    // 9) pooled mean over S -> tail of d_out
    const size_t logits_elems = (size_t)T * V;
    mean_pool<<<(Bz * H) / 256, blk, 0, stream>>>(xf32, out + logits_elems, S, H);
}
